// Encoder_35665408426681
// MI455X (gfx1250) — compile-verified, matrix operand order checked by reading, not run on hardware
//
#include <hip/hip_runtime.h>
#include <math.h>

// ---------------------------------------------------------------------------
// Problem constants (from the reference)
// ---------------------------------------------------------------------------
constexpr int NN   = 16000;   // nodes
constexpr int EE   = 64000;   // edges
constexpr int BB   = 100;     // graphs
constexpr int FIN  = 32;      // input features
constexpr int D    = 32;      // node dim
constexpr int ED   = 5;       // edge dim
constexpr int H    = 128;     // MLP hidden
constexpr int NPER = NN / BB; // nodes per graph (sorted, equal segments)
constexpr int NW2  = H * D * D; // 131072 elements in each w2

typedef __attribute__((ext_vector_type(16))) _Float16 v16h;
typedef __attribute__((ext_vector_type(8)))  _Float16 v8h;
typedef __attribute__((ext_vector_type(8)))  float    v8f;

__device__ __forceinline__ float sigf(float v) { return 1.0f / (1.0f + expf(-v)); }

// ---------------------------------------------------------------------------
// w2 pre-swizzle: f32 [128][1024] -> f16 in the exact per-lane WMMA B order.
// B is fed as B^T tiles in the documented 16-bit A layout:
//   element m of lane l, ktile t, chunk c  <->  w2[kg][col]
//   kg  = t*32 + (m>>3)*16 + 8*(l>=16) + (m&7)
//   col = (c>>1)*32 + (c&1)*16 + (l&15)        (col = i*32 + j of W_e[i][j])
// ---------------------------------------------------------------------------
__global__ void prep_w2_k(const float* __restrict__ w2, _Float16* __restrict__ out)
{
    int idx = blockIdx.x * blockDim.x + threadIdx.x;  // 64 chunks * 4 ktiles * 32 lanes
    if (idx >= 64 * 4 * 32) return;
    int lane = idx & 31;
    int t    = (idx >> 5) & 3;
    int c    = idx >> 7;
    int hi8  = (lane & 16) ? 8 : 0;
    int col  = (c >> 1) * 32 + (c & 1) * 16 + (lane & 15);
#pragma unroll
    for (int m = 0; m < 16; ++m) {
        int kg = t * 32 + ((m >> 3) * 16) + hi8 + (m & 7);
        out[(size_t)idx * 16 + m] = (_Float16)w2[(size_t)kg * (D * D) + col];
    }
}

__global__ void fill0_k(float* __restrict__ p, int n)
{
    int i = blockIdx.x * blockDim.x + threadIdx.x;
    if (i < n) p[i] = 0.0f;
}

__global__ void deg_k(const int* __restrict__ dst, float* __restrict__ cnt, int e)
{
    int i = blockIdx.x * blockDim.x + threadIdx.x;
    if (i < e) atomicAdd(&cnt[dst[i]], 1.0f);
}

// ---------------------------------------------------------------------------
// lin0: h = relu(x @ w0 + b0)
// ---------------------------------------------------------------------------
__global__ __launch_bounds__(128)
void lin0_k(const float* __restrict__ x, const float* __restrict__ w0,
            const float* __restrict__ b0, float* __restrict__ h, int n)
{
    __shared__ float ws[FIN * D];
    __shared__ float bs[D];
    int tid = threadIdx.x;
    for (int i = tid; i < FIN * D; i += 128) ws[i] = w0[i];
    if (tid < D) bs[tid] = b0[tid];
    __syncthreads();
    int node = blockIdx.x * 128 + tid;
    if (node >= n) return;
    float xi[FIN];
#pragma unroll
    for (int k = 0; k < FIN; ++k) xi[k] = x[(size_t)node * FIN + k];
#pragma unroll 4
    for (int j = 0; j < D; ++j) {
        float acc = bs[j];
#pragma unroll
        for (int k = 0; k < FIN; ++k) acc += xi[k] * ws[k * D + j];
        h[(size_t)node * D + j] = fmaxf(acc, 0.0f);
    }
}

// ---------------------------------------------------------------------------
// Fused NNConv, 32 edges per wave (two 16-row A tiles share each B chunk):
//   h_e = relu(edge_attr @ w1 + b1)        (VALU -> LDS f16, A layout source)
//   W_e chunk = h_e @ w2 + b2              (v_wmma_f32_16x16x32_f16, C = b2)
//   msg_e    += x[src_e] . W_e chunk       (VALU FMA, x tile in LDS)
//   msgsum[dst_e] += msg_e                 (global_atomic_add_f32 scatter-mean)
// W_e (256 MB/MLP if materialized) never leaves VGPRs; each B chunk-load
// feeds 8 WMMAs (2 independent accumulate chains) to halve the L2 B stream.
// ---------------------------------------------------------------------------
__global__ __launch_bounds__(128)
void nnconv_fused_k(const float* __restrict__ edge_attr,
                    const int* __restrict__ src, const int* __restrict__ dst,
                    const float* __restrict__ xn,
                    const float* __restrict__ w1, const float* __restrict__ b1,
                    const _Float16* __restrict__ w2swz,
                    const float* __restrict__ b2,
                    float* __restrict__ msgsum, int e_total)
{
    __shared__ float w1s[ED][H];
    __shared__ float b1s[H];
    __shared__ float b2s[D * D];
    __shared__ __align__(16) _Float16 hA[4][32][H];  // 4 waves x 32 edges x K=128
    __shared__ float xs[4][32][D];                   // gathered x[src] per tile
    __shared__ int   dsts[4][32];

    const int tid = threadIdx.x;
    for (int i = tid; i < ED * H; i += 128) w1s[i / H][i % H] = w1[i];
    for (int i = tid; i < H; i += 128) b1s[i] = b1[i];
    for (int i = tid; i < D * D; i += 128) b2s[i] = b2[i];
    __syncthreads();

    const int wave = tid >> 5;
    const int lane = tid & 31;
    const int tile = blockIdx.x * 4 + wave;   // 32-edge tile
    const int eb   = tile * 32;

    // ---- Phase A: one lane per edge builds h (f16) and gathers x[src] ----
    int e = eb + lane; if (e >= e_total) e = e_total - 1;
    {
        float ea[ED];
#pragma unroll
        for (int d = 0; d < ED; ++d) ea[d] = edge_attr[(size_t)e * ED + d];
        dsts[wave][lane] = dst[e];
        for (int k = 0; k < H; ++k) {
            float acc = b1s[k];
#pragma unroll
            for (int d = 0; d < ED; ++d) acc += ea[d] * w1s[d][k];
            hA[wave][lane][k] = (_Float16)fmaxf(acc, 0.0f);
        }
        int s = src[e];
        const float4* xr4 = (const float4*)(xn + (size_t)s * D);
        float4* xd4 = (float4*)&xs[wave][lane][0];
#pragma unroll
        for (int q = 0; q < D / 4; ++q) xd4[q] = xr4[q];
    }
    __syncthreads();

    // ---- Phase B: two A row tiles (documented 16-bit A layout) ----
    const int r   = lane & 15;     // output column / A row within tile
    const int hi  = lane >> 4;
    const int hi8 = hi * 8;
    v16h A[2][4];
#pragma unroll
    for (int rt = 0; rt < 2; ++rt)
#pragma unroll
        for (int t = 0; t < 4; ++t) {
            v8h lo = *(const v8h*)&hA[wave][rt * 16 + r][t * 32 + hi8];
            v8h hh = *(const v8h*)&hA[wave][rt * 16 + r][t * 32 + 16 + hi8];
            A[rt][t] = __builtin_shufflevector(lo, hh, 0, 1, 2, 3, 4, 5, 6, 7,
                                                       8, 9, 10, 11, 12, 13, 14, 15);
        }

    float macc[2][2][8];   // [rowtile][jh][v]
#pragma unroll
    for (int rt = 0; rt < 2; ++rt)
#pragma unroll
        for (int jh = 0; jh < 2; ++jh)
#pragma unroll
            for (int v = 0; v < 8; ++v) macc[rt][jh][v] = 0.0f;

    // 64 column chunks of W_e: chunk c covers cols i*32 + jh*16 + (0..15)
#pragma unroll 1
    for (int i = 0; i < D; ++i) {
        float xv[2][8];
#pragma unroll
        for (int rt = 0; rt < 2; ++rt)
#pragma unroll
            for (int v = 0; v < 8; ++v) xv[rt][v] = xs[wave][rt * 16 + v + hi8][i];
        if (i + 1 < D)   // keep next B stream hot (global_prefetch_b8)
            __builtin_prefetch(w2swz + ((size_t)((i + 1) * 2 * 4) * 32 + lane) * 16, 0, 0);
#pragma unroll
        for (int jh = 0; jh < 2; ++jh) {
            const int c = i * 2 + jh;
            const float b2v = b2s[i * D + jh * 16 + r];   // bias folded into C
            v8f a0 = {b2v, b2v, b2v, b2v, b2v, b2v, b2v, b2v};
            v8f a1 = a0;
#pragma unroll
            for (int t = 0; t < 4; ++t) {
                const v16h Bt = *(const v16h*)(w2swz + ((size_t)(c * 4 + t) * 32 + lane) * 16);
                a0 = __builtin_amdgcn_wmma_f32_16x16x32_f16(
                    false, A[0][t], false, Bt, (short)0, a0, false, false);
                a1 = __builtin_amdgcn_wmma_f32_16x16x32_f16(
                    false, A[1][t], false, Bt, (short)0, a1, false, false);
            }
            // msg_e[j] += x_src[e][i] * W_e[i][j]  (C/D layout: row = v+8*hi, col = r)
#pragma unroll
            for (int v = 0; v < 8; ++v) macc[0][jh][v] += xv[0][v] * a0[v];
#pragma unroll
            for (int v = 0; v < 8; ++v) macc[1][jh][v] += xv[1][v] * a1[v];
        }
    }

    // ---- scatter-mean numerator ----
#pragma unroll
    for (int rt = 0; rt < 2; ++rt)
#pragma unroll
        for (int v = 0; v < 8; ++v) {
            const int row = rt * 16 + v + hi8;
            if (eb + row < e_total) {
                const int dn = dsts[wave][row];
                atomicAdd(&msgsum[(size_t)dn * D + r],      macc[rt][0][v]);
                atomicAdd(&msgsum[(size_t)dn * D + 16 + r], macc[rt][1][v]);
            }
        }
}

__global__ void norm_relu_k(const float* __restrict__ msum, const float* __restrict__ cnt,
                            float* __restrict__ out, int total)
{
    int i = blockIdx.x * blockDim.x + threadIdx.x;
    if (i >= total) return;
    float c = cnt[i >> 5];
    out[i] = fmaxf(msum[i] / fmaxf(c, 1.0f), 0.0f);
}

// ---------------------------------------------------------------------------
// GRU step (gate order r,z,n), one thread per node, h updated in place
// ---------------------------------------------------------------------------
__global__ __launch_bounds__(128)
void gru_k(const float* __restrict__ m, float* __restrict__ h,
           const float* __restrict__ wih, const float* __restrict__ whh,
           const float* __restrict__ bih, const float* __restrict__ bhh, int n)
{
    __shared__ float wi[3 * D * D];
    __shared__ float wh[3 * D * D];
    __shared__ float bi[3 * D];
    __shared__ float bh[3 * D];
    int tid = threadIdx.x;
    for (int i = tid; i < 3 * D * D; i += 128) { wi[i] = wih[i]; wh[i] = whh[i]; }
    for (int i = tid; i < 3 * D; i += 128)     { bi[i] = bih[i]; bh[i] = bhh[i]; }
    __syncthreads();
    int node = blockIdx.x * 128 + tid;
    if (node >= n) return;
    float xv[D], hv[D], hn2[D];
#pragma unroll
    for (int k = 0; k < D; ++k) { xv[k] = m[(size_t)node * D + k]; hv[k] = h[(size_t)node * D + k]; }
    for (int j = 0; j < D; ++j) {
        float ir = bi[j], iz = bi[D + j], in2 = bi[2 * D + j];
        float hr = bh[j], hz = bh[D + j], hn = bh[2 * D + j];
#pragma unroll
        for (int k = 0; k < D; ++k) {
            ir  += xv[k] * wi[j * D + k];
            iz  += xv[k] * wi[(D + j) * D + k];
            in2 += xv[k] * wi[(2 * D + j) * D + k];
            hr  += hv[k] * wh[j * D + k];
            hz  += hv[k] * wh[(D + j) * D + k];
            hn  += hv[k] * wh[(2 * D + j) * D + k];
        }
        float rr = sigf(ir + hr);
        float zz = sigf(iz + hz);
        float nn = tanhf(in2 + rr * hn);
        hn2[j] = (1.0f - zz) * nn + zz * hv[j];
    }
#pragma unroll
    for (int j = 0; j < D; ++j) h[(size_t)node * D + j] = hn2[j];
}

// ---------------------------------------------------------------------------
// Set2Set: one block per graph; segments contiguous, NPER nodes each.
// LSTM(2D->D), gate order (i,f,g,o); 3 steps; output q_star [B, 2D].
// ---------------------------------------------------------------------------
__global__ __launch_bounds__(192)
void set2set_k(const float* __restrict__ xn,
               const float* __restrict__ wih, const float* __restrict__ whh,
               const float* __restrict__ bih, const float* __restrict__ bhh,
               float* __restrict__ qout, int nper)
{
    __shared__ float wi[4 * D * 2 * D];   // 128 x 64
    __shared__ float wh[4 * D * D];       // 128 x 32
    __shared__ float bb[4 * D];
    __shared__ float hs[D], cs[D], qstar[2 * D], gl[4 * D];
    __shared__ float el[NPER];
    __shared__ float red[2];

    const int tid = threadIdx.x;
    const int base = blockIdx.x * nper;
    for (int i = tid; i < 4 * D * 2 * D; i += 192) wi[i] = wih[i];
    for (int i = tid; i < 4 * D * D; i += 192)     wh[i] = whh[i];
    for (int i = tid; i < 4 * D; i += 192)         bb[i] = bih[i] + bhh[i];
    if (tid < D) { hs[tid] = 0.0f; cs[tid] = 0.0f; }
    if (tid < 2 * D) qstar[tid] = 0.0f;
    __syncthreads();

    for (int step = 0; step < 3; ++step) {
        if (tid < 4 * D) {
            float g = bb[tid];
            for (int k = 0; k < 2 * D; ++k) g += qstar[k] * wi[tid * 2 * D + k];
            for (int k = 0; k < D; ++k)     g += hs[k] * wh[tid * D + k];
            gl[tid] = g;
        }
        __syncthreads();
        if (tid < D) {
            float ig = gl[tid], fg = gl[D + tid], gg = gl[2 * D + tid], og = gl[3 * D + tid];
            float cN = sigf(fg) * cs[tid] + sigf(ig) * tanhf(gg);
            cs[tid] = cN;
            hs[tid] = sigf(og) * tanhf(cN);
        }
        __syncthreads();
        for (int nl = tid; nl < nper; nl += 192) {
            float acc = 0.0f;
            const float* xr = xn + (size_t)(base + nl) * D;
            for (int k = 0; k < D; ++k) acc += xr[k] * hs[k];
            el[nl] = acc;
        }
        __syncthreads();
        if (tid == 0) {
            float mx = -3.4e38f;
            for (int nl = 0; nl < nper; ++nl) mx = fmaxf(mx, el[nl]);
            red[0] = mx;
        }
        __syncthreads();
        float mx = red[0];
        for (int nl = tid; nl < nper; nl += 192) el[nl] = expf(el[nl] - mx);
        __syncthreads();
        if (tid == 0) {
            float s = 0.0f;
            for (int nl = 0; nl < nper; ++nl) s += el[nl];
            red[1] = s;
        }
        __syncthreads();
        float s = red[1];
        if (tid < D) {
            float acc = 0.0f;
            for (int nl = 0; nl < nper; ++nl)
                acc += el[nl] * xn[(size_t)(base + nl) * D + tid];
            qstar[D + tid] = acc / s;
            qstar[tid] = hs[tid];
        }
        __syncthreads();
    }
    for (int k = tid; k < 2 * D; k += 192) qout[(size_t)blockIdx.x * 2 * D + k] = qstar[k];
}

__global__ void expand_k(const int* __restrict__ batch,
                         const float* __restrict__ gmu, const float* __restrict__ glv,
                         float* __restrict__ omu, float* __restrict__ olv, int n)
{
    int i = blockIdx.x * blockDim.x + threadIdx.x;
    if (i >= n * 2 * D) return;
    int node = i >> 6, k = i & 63;
    int b = batch[node];
    omu[i] = gmu[(size_t)b * 2 * D + k];
    olv[i] = glv[(size_t)b * 2 * D + k];
}

// ---------------------------------------------------------------------------
// Orchestration.  Input flattening (setup_inputs dict order, tuples in order):
//  0 x  1 edge_attr  2 lin0.w0  3 lin0.b0
//  4..7   mlp_conv (w1,b1,w2,b2)        8..11  gru (w_ih,w_hh,b_ih,b_hh)
// 12..15 mlp_node_mu  16..19 mlp_node_lv  20..23 mlp_graph_mu  24..27 mlp_graph_lv
// 28..31 s2s_nodes   32..35 s2s_mu   36..39 s2s_lv
// 40 edge_index [2,E]  41 batch [N]
// d_out: node_mu | node_lv | grouped_mu_exp | grouped_lv_exp | node_graph
// ---------------------------------------------------------------------------
extern "C" void kernel_launch(void* const* d_in, const int* in_sizes, int n_in,
                              void* d_out, int out_size, void* d_ws, size_t ws_size,
                              hipStream_t stream)
{
    (void)in_sizes; (void)n_in; (void)out_size; (void)ws_size;

    const float* x    = (const float*)d_in[0];
    const float* ea   = (const float*)d_in[1];
    const float* l0w  = (const float*)d_in[2];
    const float* l0b  = (const float*)d_in[3];
    const int*   eidx = (const int*)d_in[40];
    const int*   bat  = (const int*)d_in[41];
    const int* srcp = eidx;
    const int* dstp = eidx + EE;
    const int mlpIdx[5] = {4, 12, 16, 20, 24};  // conv, node_mu, node_lv, graph_mu, graph_lv

    // workspace layout: f16 swizzled w2 x5, then f32 buffers
    char* wsb = (char*)d_ws;
    _Float16* w2hBase = (_Float16*)wsb;
    float* f = (float*)(wsb + (size_t)5 * NW2 * sizeof(_Float16));
    float* hbuf = f; f += (size_t)NN * D;
    float* msg  = f; f += (size_t)NN * D;
    float* mbuf = f; f += (size_t)NN * D;
    float* cnt  = f; f += NN;
    float* gmu  = f; f += (size_t)NN * D;
    float* glv  = f; f += (size_t)NN * D;
    float* qmu  = f; f += BB * 2 * D;
    float* qlv  = f; f += BB * 2 * D;

    // 1) swizzle all five w2 matrices to f16 WMMA-B order (L2-resident streams)
    for (int mi = 0; mi < 5; ++mi)
        prep_w2_k<<<(64 * 4 * 32 + 255) / 256, 256, 0, stream>>>(
            (const float*)d_in[mlpIdx[mi] + 2], w2hBase + (size_t)mi * NW2);

    // 2) in-degrees for scatter-mean
    fill0_k<<<(NN + 255) / 256, 256, 0, stream>>>(cnt, NN);
    deg_k<<<(EE + 255) / 256, 256, 0, stream>>>(dstp, cnt, EE);

    // 3) lin0 -> h (also the GRU hidden init)
    lin0_k<<<(NN + 127) / 128, 128, 0, stream>>>(x, l0w, l0b, hbuf, NN);

    // 4) three message-passing + GRU steps (fused NNConv, conv MLP)
    for (int t = 0; t < 3; ++t) {
        fill0_k<<<(NN * D + 255) / 256, 256, 0, stream>>>(msg, NN * D);
        nnconv_fused_k<<<(EE + 127) / 128, 128, 0, stream>>>(
            ea, srcp, dstp, hbuf,
            (const float*)d_in[4], (const float*)d_in[5],
            w2hBase, (const float*)d_in[7], msg, EE);
        norm_relu_k<<<(NN * D + 255) / 256, 256, 0, stream>>>(msg, cnt, mbuf, NN * D);
        gru_k<<<(NN + 127) / 128, 128, 0, stream>>>(
            mbuf, hbuf,
            (const float*)d_in[8], (const float*)d_in[9],
            (const float*)d_in[10], (const float*)d_in[11], NN);
    }

    float* out = (float*)d_out;
    float* out_node_mu = out;
    float* out_node_lv = out + (size_t)NN * D;
    float* out_gmu_exp = out + (size_t)2 * NN * D;
    float* out_glv_exp = out + (size_t)2 * NN * D + (size_t)NN * 2 * D;
    float* out_ngraph  = out + (size_t)2 * NN * D + (size_t)2 * NN * 2 * D;

    // 5) node_graph = Set2Set(out)
    set2set_k<<<BB, 192, 0, stream>>>(
        hbuf, (const float*)d_in[28], (const float*)d_in[29],
        (const float*)d_in[30], (const float*)d_in[31], out_ngraph, NPER);

    // 6) four NNConv heads (fused), node heads straight into d_out
    float* headDst[4] = {out_node_mu, out_node_lv, gmu, glv};
    for (int hh = 0; hh < 4; ++hh) {
        int bi = mlpIdx[hh + 1];
        fill0_k<<<(NN * D + 255) / 256, 256, 0, stream>>>(msg, NN * D);
        nnconv_fused_k<<<(EE + 127) / 128, 128, 0, stream>>>(
            ea, srcp, dstp, hbuf,
            (const float*)d_in[bi], (const float*)d_in[bi + 1],
            w2hBase + (size_t)(hh + 1) * NW2, (const float*)d_in[bi + 3], msg, EE);
        norm_relu_k<<<(NN * D + 255) / 256, 256, 0, stream>>>(msg, cnt, headDst[hh], NN * D);
    }

    // 7) graph-level Set2Sets + expansion via batch gather
    set2set_k<<<BB, 192, 0, stream>>>(
        gmu, (const float*)d_in[32], (const float*)d_in[33],
        (const float*)d_in[34], (const float*)d_in[35], qmu, NPER);
    set2set_k<<<BB, 192, 0, stream>>>(
        glv, (const float*)d_in[36], (const float*)d_in[37],
        (const float*)d_in[38], (const float*)d_in[39], qlv, NPER);
    expand_k<<<(NN * 2 * D + 255) / 256, 256, 0, stream>>>(
        bat, qmu, qlv, out_gmu_exp, out_glv_exp, NN);
}
